// NetWave_InN_legendre_80865644249643
// MI455X (gfx1250) — matrix operand order repaired and checked by reading, not compiled, not run on hardware
//
#include <hip/hip_runtime.h>
#include <hip/hip_bf16.h>

typedef __attribute__((ext_vector_type(16))) _Float16 v16h;
typedef __attribute__((ext_vector_type(8)))  _Float16 v8h;
typedef __attribute__((ext_vector_type(8)))  float    v8f;

#define CH 40
#define CPAD 64   // padded input channels for WMMA K
#define COPAD 48  // padded output channels (3 n-tiles of 16)
#define MODES 16

__device__ __forceinline__ float gelu_f(float v) {
  return 0.5f * v * (1.0f + erff(v * 0.70710678118654752440f));
}

// ---------------- pad (wrap) + prescale ch1 by 0.1 ----------------
__global__ void k_pad(const float* __restrict__ x, float* __restrict__ P) {
  const int HP = 242, WP = 242, Hs = 190, Ws = 190, R = 25;
  long i = (long)blockIdx.x * blockDim.x + threadIdx.x;
  long total = 8L * 2 * HP * WP;
  if (i >= total) return;
  int xx = (int)(i % WP); long r = i / WP;
  int yy = (int)(r % HP); r /= HP;
  int ch = (int)(r % 2);  int b = (int)(r / 2);
  int sy = (yy - R) % Hs; if (sy < 0) sy += Hs;
  int sx = (xx - R) % Ws; if (sx < 0) sx += Ws;
  float v = x[(((long)b * 2 + ch) * Hs + sy) * Ws + sx];
  if (ch == 1) v *= 0.1f;
  P[i] = v;
}

// ---------------- first conv 3x3, 2 -> 40 (K=18, too small for WMMA) --------------
__global__ void k_conv_first(const float* __restrict__ P, const float* __restrict__ w,
                             float* __restrict__ out) {
  const int LP = 242, LO = 240;
  long i = (long)blockIdx.x * blockDim.x + threadIdx.x;
  long total = 8L * CH * LO * LO;
  if (i >= total) return;
  int x = (int)(i % LO); long r = i / LO;
  int y = (int)(r % LO); r /= LO;
  int co = (int)(r % CH); int b = (int)(r / CH);
  float acc = 0.f;
  for (int ci = 0; ci < 2; ++ci)
    for (int ky = 0; ky < 3; ++ky)
      for (int kx = 0; kx < 3; ++kx)
        acc += P[(((long)b * 2 + ci) * LP + (y + ky)) * LP + (x + kx)] *
               w[((co * 2 + ci) * 3 + ky) * 3 + kx];
  out[i] = acc;
}

// ------- fp32 NCHW -> fp16 NHWC with channel pad 40->64 (pad lanes zeroed) --------
__global__ void k_tofp16_nhwc(const float* __restrict__ in, _Float16* __restrict__ o16,
                              int L) {
  long i = (long)blockIdx.x * blockDim.x + threadIdx.x;
  long total = 8L * L * L * CPAD;
  if (i >= total) return;
  int c = (int)(i % CPAD); long r = i / CPAD;
  int x = (int)(r % L); r /= L;
  int y = (int)(r % L); int b = (int)(r / L);
  float v = (c < CH) ? in[(((size_t)b * CH + c) * L + y) * L + x] : 0.f;
  o16[i] = (_Float16)v;
}

// ------- weights fp32 (8 layers of (40,40,3,3)) -> fp16 [layer][tap][co48][ci64] ---
__global__ void k_prep_weights(const float* __restrict__ cA, const float* __restrict__ cB,
                               _Float16* __restrict__ w16) {
  long i = (long)blockIdx.x * blockDim.x + threadIdx.x;
  long total = 8L * 9 * COPAD * CPAD;
  if (i >= total) return;
  int ci = (int)(i % CPAD); long r = i / CPAD;
  int co = (int)(r % COPAD); r /= COPAD;
  int tap = (int)(r % 9); int layer = (int)(r / 9);
  float v = 0.f;
  if (co < CH && ci < CH) {
    const float* src = (layer < 4) ? (cA + (size_t)layer * CH * CH * 9)
                                   : (cB + (size_t)(layer - 4) * CH * CH * 9);
    v = src[((size_t)co * CH + ci) * 9 + tap];
  }
  w16[i] = (_Float16)v;
}

// ---- 3x3 40->40 conv: one wave = 16 positions x 48 channels (3 accumulators) -----
// 9 tap-GEMMs, K=64 (2 steps) each, A loaded once and reused by 3 n-tiles:
// 54 x V_WMMA_F32_16X16X32_F16 per wave.
// A (16x32 f16, ISA 7.12.2): lane half h -> k = ks*32 + h*16 + [0..15]:
//   two contiguous b128 loads from fp16 NHWC.
// B (32x16 f16, ISA 7.12.4): lane half h -> k = ks*32 + h*16 + [0..15]:
//   one aligned 32B v16h load; tap/nt/ks offsets fold to immediates.
__global__ void k_conv3x3_wmma(const _Float16* __restrict__ inh,
                               const _Float16* __restrict__ wh,
                               float* __restrict__ outF32, _Float16* __restrict__ outF16,
                               int L, int Lout, int mTiles, int totalWaves, int doGelu) {
  int wid = blockIdx.x * (blockDim.x >> 5) + (threadIdx.x >> 5);
  if (wid >= totalWaves) return;  // wave-uniform exit: EXEC all-1s at every WMMA
  int mt = wid % mTiles; int b = wid / mTiles;
  int lane = threadIdx.x & 31;
  int half = lane >> 4;
  int l16  = lane & 15;
  int MM = Lout * Lout;
  int mA = mt * 16 + l16;
  int pos = (mA < MM) ? mA : (MM - 1);
  int ay = pos / Lout, ax = pos % Lout;
  const _Float16* ibase = inh + (size_t)b * L * L * CPAD;
  // per-lane B base: row l16, +half*16 within k-step; tap/nt/ks are immediates
  const _Float16* wb = wh + (size_t)l16 * CPAD + half * 16;
  v8f acc0 = {0.f, 0.f, 0.f, 0.f, 0.f, 0.f, 0.f, 0.f};
  v8f acc1 = acc0, acc2 = acc0;
#pragma unroll
  for (int ky = 0; ky < 3; ++ky) {
#pragma unroll
    for (int kx = 0; kx < 3; ++kx) {
      const int tap = ky * 3 + kx;
      const _Float16* pA = ibase + ((size_t)(ay + ky) * L + (ax + kx)) * CPAD;
#pragma unroll
      for (int ks = 0; ks < 2; ++ks) {
        v8h a0 = *(const v8h*)(pA + ks * 32 + half * 16);
        v8h a1 = *(const v8h*)(pA + ks * 32 + half * 16 + 8);
        v16h a = __builtin_shufflevector(a0, a1, 0, 1, 2, 3, 4, 5, 6, 7,
                                         8, 9, 10, 11, 12, 13, 14, 15);
        v16h b0 = *(const v16h*)(wb + ((size_t)tap * COPAD +  0) * CPAD + ks * 32);
        v16h b1 = *(const v16h*)(wb + ((size_t)tap * COPAD + 16) * CPAD + ks * 32);
        v16h b2 = *(const v16h*)(wb + ((size_t)tap * COPAD + 32) * CPAD + ks * 32);
        acc0 = __builtin_amdgcn_wmma_f32_16x16x32_f16(false, a, false, b0,
                                                      (short)0, acc0, false, false);
        acc1 = __builtin_amdgcn_wmma_f32_16x16x32_f16(false, a, false, b1,
                                                      (short)0, acc1, false, false);
        acc2 = __builtin_amdgcn_wmma_f32_16x16x32_f16(false, a, false, b2,
                                                      (short)0, acc2, false, false);
      }
    }
  }
#pragma unroll
  for (int v = 0; v < 8; ++v) {
    int m = mt * 16 + v + half * 8;   // C/D layout: VGPR v -> M=v (+8 hi lanes)
    if (m < MM) {
      int oy = m / Lout, ox = m % Lout;
      float v0 = acc0[v], v1 = acc1[v], v2 = acc2[v];
      if (doGelu) { v0 = gelu_f(v0); v1 = gelu_f(v1); v2 = gelu_f(v2); }
      if (outF16) {  // fused fp16-NHWC restage (t1 path, only consumed by convB)
        _Float16* o = outF16 + (((size_t)b * Lout + oy) * Lout + ox) * CPAD + l16;
        o[0]  = (_Float16)v0;
        o[16] = (_Float16)v1;
        if (l16 < 8) o[32] = (_Float16)v2;
      } else {
        float* o = outF32 + ((size_t)b * CH * Lout + (size_t)oy) * Lout + ox;
        size_t cs = (size_t)Lout * Lout;
        o[(size_t)l16 * cs]        = v0;
        o[(size_t)(16 + l16) * cs] = v1;
        if (l16 < 8) o[(size_t)(32 + l16) * cs] = v2;
      }
    }
  }
}

// ---------------- Legendre decompose: 12x12 stride-6, 16 modes (x0.5 folded) ------
__global__ void k_decompose(const float* __restrict__ xin, const float* __restrict__ fd,
                            float* __restrict__ Lm, int L, int ll) {
  long i = (long)blockIdx.x * blockDim.x + threadIdx.x;
  long total = 8L * CH * ll * ll;
  if (i >= total) return;
  int q = (int)(i % ll); long r = i / ll;
  int p = (int)(r % ll); r /= ll;
  int ch = (int)(r % CH); int b = (int)(r / CH);
  const float* img = xin + ((size_t)b * CH + ch) * L * L;
  float acc[MODES];
#pragma unroll
  for (int m = 0; m < MODES; ++m) acc[m] = 0.f;
  for (int u = 0; u < 12; ++u)
    for (int v = 0; v < 12; ++v) {
      float px = img[(size_t)(p * 6 + u) * L + (q * 6 + v)];
#pragma unroll
      for (int m = 0; m < MODES; ++m) acc[m] += px * fd[m * 144 + u * 12 + v];
    }
#pragma unroll
  for (int m = 0; m < MODES; ++m)
    Lm[(((size_t)b * (MODES * CH) + m * CH + ch) * ll + p) * ll + q] = 0.5f * acc[m];
}

// ---------------- grouped 1x1 mode mix: block-diag 40 x (16x16), in-place, x2 -----
__global__ void k_modemix(float* __restrict__ Lm, const float* __restrict__ Wm, int ll) {
  long i = (long)blockIdx.x * blockDim.x + threadIdx.x;
  long total = 8L * CH * ll * ll;  // (b, group, pos)
  if (i >= total) return;
  int pos = (int)(i % (ll * ll)); long r = i / (ll * ll);
  int g = (int)(r % CH); int b = (int)(r / CH);
  size_t stride = (size_t)ll * ll;
  size_t base = ((size_t)b * (MODES * CH) + g * 16) * stride + pos;
  float v[16];
#pragma unroll
  for (int k = 0; k < 16; ++k) v[k] = Lm[base + k * stride];
#pragma unroll
  for (int j = 0; j < 16; ++j) {
    float s = 0.f;
#pragma unroll
    for (int k = 0; k < 16; ++k) s += Wm[(g * 16 + j) * 16 + k] * v[k];
    Lm[base + j * stride] = 2.0f * s;
  }
}

// -------- conv_transpose stride-6 reconstruct + crop(6) + residual add + GELU -----
__global__ void k_recon_combine(const float* __restrict__ Lm, const float* __restrict__ fr,
                                const float* __restrict__ lin, float* __restrict__ xo,
                                int ll, int Ln, int Llin) {
  long i = (long)blockIdx.x * blockDim.x + threadIdx.x;
  long total = 8L * CH * Ln * Ln;
  if (i >= total) return;
  int j = (int)(i % Ln); long r = i / Ln;
  int ii = (int)(r % Ln); r /= Ln;
  int ch = (int)(r % CH); int b = (int)(r / CH);
  int oi = ii + 6, oj = j + 6;
  int u0 = oi % 6, p0 = oi / 6;   // interior crop: both taps always valid
  int v0 = oj % 6, q0 = oj / 6;
  float acc = 0.f;
  const float* LB = Lm + (size_t)b * (MODES * CH) * ll * ll;
#pragma unroll
  for (int du = 0; du < 2; ++du) {
    int u = u0 + du * 6; int p = p0 - du;
#pragma unroll
    for (int dv = 0; dv < 2; ++dv) {
      int v = v0 + dv * 6; int q = q0 - dv;
      size_t pidx = (size_t)p * ll + q;
#pragma unroll
      for (int m = 0; m < MODES; ++m)
        acc += LB[((size_t)(m * CH + ch)) * ll * ll + pidx] * fr[m * 144 + u * 12 + v];
    }
  }
  acc *= 0.5f;  // filter_r / K
  acc += lin[(((size_t)b * CH + ch) * Llin + (ii + 4)) * Llin + (j + 4)];
  xo[i] = gelu_f(acc);
}

// ---------------- head: 1x1 40->128 + GELU ----------------
__global__ void k_head1(const float* __restrict__ xin, const float* __restrict__ w11,
                        float* __restrict__ Hb) {
  const int L = 192;
  long i = (long)blockIdx.x * blockDim.x + threadIdx.x;
  long total = 8L * 128 * L * L;
  if (i >= total) return;
  int pos = (int)(i % (L * L)); long r = i / (L * L);
  int co = (int)(r % 128); int b = (int)(r / 128);
  const float* xb = xin + (size_t)b * CH * L * L;
  float acc = 0.f;
  for (int ci = 0; ci < CH; ++ci)
    acc += xb[(size_t)ci * L * L + pos] * w11[co * CH + ci];
  Hb[i] = gelu_f(acc);
}

// ---------------- head: 1x1 128->2, x10 on ch1, crop to 190 ----------------
__global__ void k_head2(const float* __restrict__ Hb, const float* __restrict__ wout,
                        float* __restrict__ out) {
  const int L = 192, LO = 190;
  long i = (long)blockIdx.x * blockDim.x + threadIdx.x;
  long total = 8L * 2 * LO * LO;
  if (i >= total) return;
  int x = (int)(i % LO); long r = i / LO;
  int y = (int)(r % LO); r /= LO;
  int oc = (int)(r % 2); int b = (int)(r / 2);
  const float* hb = Hb + (size_t)b * 128 * L * L;
  float acc = 0.f;
  for (int co = 0; co < 128; ++co)
    acc += hb[(size_t)co * L * L + (size_t)y * L + x] * wout[oc * 128 + co];
  if (oc == 1) acc *= 10.f;
  out[i] = acc;
}

extern "C" void kernel_launch(void* const* d_in, const int* in_sizes, int n_in,
                              void* d_out, int out_size, void* d_ws, size_t ws_size,
                              hipStream_t stream) {
  (void)in_sizes; (void)n_in; (void)out_size; (void)ws_size;
  const float* x      = (const float*)d_in[0];
  const float* wfirst = (const float*)d_in[1];
  const float* convA  = (const float*)d_in[2];
  const float* convB  = (const float*)d_in[3];
  const float* Wmode  = (const float*)d_in[4];
  const float* w11    = (const float*)d_in[5];
  const float* wout   = (const float*)d_in[6];
  const float* fd     = (const float*)d_in[7];
  const float* fr     = (const float*)d_in[8];
  float* out = (float*)d_out;
  float* ws  = (float*)d_ws;

  // workspace (fp32-element offsets, ~370 MB total):
  float* P   = ws;                   // 8*2*242*242            =   937024
  float* A   = P   + 937024;         // 8*40*240*240           = 18432000
  float* Bb  = A   + 18432000;       // 8*40*238*238           = 18126080
  float* Cc  = Bb  + 18126080;       // 8*40*236*236           = 17822720
  float* Lm  = Cc  + 17822720;       // 8*640*39*39            =  7787520
  float* F0  = Lm  + 7787520;        // fp16 NHWC x:  8*240*240*64 halfs = 14745600 fl
  float* F1  = F0  + 14745600;       // fp16 NHWC t1: 8*238*238*64 halfs = 14500864 fl
  float* W6  = F1  + 14500864;       // fp16 weights: 8*9*48*64 halfs    =   110592 fl
  _Float16* F16X = (_Float16*)F0;
  _Float16* F16T = (_Float16*)F1;
  _Float16* W16  = (_Float16*)W6;
  float* Hb = Bb;  // head buffer reuses B.. (37.7M floats available downstream)

  { long n = 8L * 9 * COPAD * CPAD;  // one-time padded fp16 weight prep (8 layers)
    k_prep_weights<<<(int)((n + 255) / 256), 256, 0, stream>>>(convA, convB, W16); }
  { long n = 8L * 2 * 242 * 242;
    k_pad<<<(int)((n + 255) / 256), 256, 0, stream>>>(x, P); }
  { long n = 8L * CH * 240 * 240;
    k_conv_first<<<(int)((n + 255) / 256), 256, 0, stream>>>(P, wfirst, A); }

  float* cur = A; float* alt = Bb;
  int L = 240;
  for (int i = 0; i < 4; ++i) {
    int L1 = L - 2, L2 = L - 4, ll = L / 6 - 1, Ln = L - 12;
    const _Float16* wA16 = W16 + (size_t)i * 9 * COPAD * CPAD;
    const _Float16* wB16 = W16 + (size_t)(4 + i) * 9 * COPAD * CPAD;
    const float* wM = Wmode + (size_t)i * MODES * CH * MODES;

    { long n = 8L * L * L * CPAD;     // restage x as fp16 NHWC (pad ch zeroed)
      k_tofp16_nhwc<<<(int)((n + 255) / 256), 256, 0, stream>>>(cur, F16X, L); }
    { int mT = (L1 * L1 + 15) / 16; int waves = 8 * mT;  // convA+GELU -> fp16 NHWC t1
      k_conv3x3_wmma<<<(waves + 3) / 4, 128, 0, stream>>>(F16X, wA16, nullptr, F16T,
                                                          L, L1, mT, waves, 1); }
    { int mT = (L2 * L2 + 15) / 16; int waves = 8 * mT;  // convB -> fp32 linear_x
      k_conv3x3_wmma<<<(waves + 3) / 4, 128, 0, stream>>>(F16T, wB16, Cc, nullptr,
                                                          L1, L2, mT, waves, 0); }
    { long n = 8L * CH * ll * ll;
      k_decompose<<<(int)((n + 255) / 256), 256, 0, stream>>>(cur, fd, Lm, L, ll); }
    { long n = 8L * CH * ll * ll;
      k_modemix<<<(int)((n + 255) / 256), 256, 0, stream>>>(Lm, wM, ll); }
    { long n = 8L * CH * Ln * Ln;     // next x written into the other fp32 buffer
      k_recon_combine<<<(int)((n + 255) / 256), 256, 0, stream>>>(Lm, fr, Cc, alt,
                                                                  ll, Ln, L2); }
    float* t = cur; cur = alt; alt = t;
    L = Ln;
  }
  // after 4 blocks: cur == A (192x192); downstream regions free for the head
  { long n = 8L * 128 * 192 * 192;
    k_head1<<<(int)((n + 255) / 256), 256, 0, stream>>>(cur, w11, Hb); }
  { long n = 8L * 2 * 190 * 190;
    k_head2<<<(int)((n + 255) / 256), 256, 0, stream>>>(Hb, wout, out); }
}
